// RoPEALiBiLatentCrossAttention_20194936226122
// MI455X (gfx1250) — compile-verified, matrix operand order checked by reading, not run on hardware
//
#include <hip/hip_runtime.h>
#include <stddef.h>

// ---------------- problem constants ----------------
constexpr int Bb = 4;
constexpr int Lq = 2048;
constexpr int Dm = 512;
constexpr int Hh = 8;
constexpr int Pp = 256;
constexpr int HD = 64;
constexpr int ML = Bb * Lq; // 8192 rows

// ---------------- WMMA vector types ----------------
typedef __bf16 v8bf  __attribute__((ext_vector_type(8)));
typedef __bf16 v16bf __attribute__((ext_vector_type(16)));
typedef float  v8f   __attribute__((ext_vector_type(8)));

static __device__ __forceinline__ v16bf cat16(v8bf a, v8bf b) {
    v16bf r;
#pragma unroll
    for (int i = 0; i < 8; ++i) { r[i] = a[i]; r[i + 8] = b[i]; }
    return r;
}
static __device__ __forceinline__ v8bf ld8(const __bf16* p) {
    return *(const v8bf*)p;
}
static __device__ __forceinline__ v16bf ldfrag(const __bf16* p) {
    return cat16(ld8(p), ld8(p + 16));
}
static __device__ __forceinline__ v8f wmma_bf16(v16bf a, v16bf b, v8f c) {
    return __builtin_amdgcn_wmma_f32_16x16x32_bf16(false, a, false, b, (short)0, c,
                                                   false, false);
}

// ---------------- elementwise helpers ----------------
__global__ __launch_bounds__(256) void cvt_bf16_kernel(const float* __restrict__ in,
                                                       __bf16* __restrict__ out, int n) {
    int i = blockIdx.x * blockDim.x + threadIdx.x;
    if (i < n) out[i] = (__bf16)in[i];
}

// W[K,N] f32 -> Wt[N,K] bf16
__global__ __launch_bounds__(256) void transpose_w_kernel(const float* __restrict__ W,
                                                          __bf16* __restrict__ Wt,
                                                          int K, int N) {
    int i = blockIdx.x * blockDim.x + threadIdx.x;
    if (i >= K * N) return;
    int n = i % N, k = i / N;
    Wt[(size_t)n * K + k] = (__bf16)W[(size_t)k * N + n];
}

// v[b*L+l, D] bf16 -> vT[((b*H+h)*HD+d)*L + l]
__global__ __launch_bounds__(256) void transpose_v_kernel(const __bf16* __restrict__ v,
                                                          __bf16* __restrict__ vT) {
    int i = blockIdx.x * blockDim.x + threadIdx.x;
    if (i >= ML * Dm) return;
    int col = i % Dm, row = i / Dm;
    int l = row % Lq, b = row / Lq;
    int h = col / HD, d = col % HD;
    vT[((size_t)(b * Hh + h) * HD + d) * Lq + l] = v[i];
}

// rotate-half RoPE on f32 [ML, D], write bf16 same layout
__global__ __launch_bounds__(256) void rope_kernel(const float* __restrict__ x,
                                                   const float* __restrict__ pos_emb,
                                                   __bf16* __restrict__ out) {
    int idx = blockIdx.x * blockDim.x + threadIdx.x; // over ML*H*(HD/2)
    if (idx >= ML * Hh * (HD / 2)) return;
    int d = idx & 31;
    int h = (idx >> 5) & (Hh - 1);
    int row = idx >> 8;            // b*L + l
    int l = row & (Lq - 1);
    float ang = pos_emb[l * (HD / 2) + d];
    float c = __cosf(ang), s = __sinf(ang);
    size_t base = (size_t)row * Dm + h * HD;
    float x1 = x[base + d], x2 = x[base + 32 + d];
    out[base + d]      = (__bf16)(x1 * c - x2 * s);
    out[base + 32 + d] = (__bf16)(x1 * s + x2 * c);
}

// ---------------- bf16 WMMA GEMM: C[M,N] = A[M,K] @ Bt[N,K]^T + bias ----------------
// one wave per 16x64 tile: one shared A-fragment, 4 B-fragments, 4 independent
// accumulation chains, software-pipelined so loads for step k overlap WMMAs of
// step k-32 (distinct live ranges -> partial s_wait_loadcnt).
__global__ __launch_bounds__(256) void gemm_bf16_kernel(const __bf16* __restrict__ A,
                                                        const __bf16* __restrict__ Bt,
                                                        const float* __restrict__ bias,
                                                        float* __restrict__ Cf,
                                                        __bf16* __restrict__ Cb,
                                                        int M, int N, int K) {
    int wave = threadIdx.x >> 5;
    int lane = threadIdx.x & 31;
    int lo = lane & 15, hi = lane >> 4, kb = hi * 8;
    int tiles_n = N >> 6;                      // 64-wide supertiles
    int gid = blockIdx.x * (blockDim.x >> 5) + wave;
    if (gid >= (M >> 4) * tiles_n) return;
    int mt = gid / tiles_n, nt = gid % tiles_n;

    const __bf16* arow = A + (size_t)(mt * 16 + lo) * K + kb;
    const __bf16* b0 = Bt + (size_t)(nt * 64 +  0 + lo) * K + kb;
    const __bf16* b1 = Bt + (size_t)(nt * 64 + 16 + lo) * K + kb;
    const __bf16* b2 = Bt + (size_t)(nt * 64 + 32 + lo) * K + kb;
    const __bf16* b3 = Bt + (size_t)(nt * 64 + 48 + lo) * K + kb;

    v8f c0 = {}, c1 = {}, c2 = {}, c3 = {};

    // prologue: fragments for k0 = 0
    v16bf a_c  = ldfrag(arow);
    v16bf m0_c = ldfrag(b0);
    v16bf m1_c = ldfrag(b1);
    v16bf m2_c = ldfrag(b2);
    v16bf m3_c = ldfrag(b3);

    for (int k0 = 32; k0 < K; k0 += 32) {
        // next-step fragments: issued before this step's WMMAs
        __builtin_prefetch(arow + k0 + 32, 0, 3);
        v16bf a_n  = ldfrag(arow + k0);
        v16bf m0_n = ldfrag(b0 + k0);
        v16bf m1_n = ldfrag(b1 + k0);
        v16bf m2_n = ldfrag(b2 + k0);
        v16bf m3_n = ldfrag(b3 + k0);

        c0 = wmma_bf16(a_c, m0_c, c0);
        c1 = wmma_bf16(a_c, m1_c, c1);
        c2 = wmma_bf16(a_c, m2_c, c2);
        c3 = wmma_bf16(a_c, m3_c, c3);

        a_c = a_n; m0_c = m0_n; m1_c = m1_n; m2_c = m2_n; m3_c = m3_n;
    }
    // epilogue step
    c0 = wmma_bf16(a_c, m0_c, c0);
    c1 = wmma_bf16(a_c, m1_c, c1);
    c2 = wmma_bf16(a_c, m2_c, c2);
    c3 = wmma_bf16(a_c, m3_c, c3);

    v8f acc[4] = {c0, c1, c2, c3};
#pragma unroll
    for (int j = 0; j < 4; ++j) {
#pragma unroll
        for (int r = 0; r < 8; ++r) {
            int m = mt * 16 + r + 8 * hi;
            int n = nt * 64 + j * 16 + lo;
            float v = acc[j][r] + bias[n];
            if (Cf) Cf[(size_t)m * N + n] = v;
            if (Cb) Cb[(size_t)m * N + n] = (__bf16)v;
        }
    }
}

// ---------------- fused flash attention ----------------
// one wave = 16 queries for one (b,h); streams 32 keys / iteration.
// K and V fragments are hoisted to the top of each iteration so the softmax
// VALU section covers the V-load latency.
__global__ __launch_bounds__(128) void attn_kernel(const __bf16* __restrict__ q,
                                                   const __bf16* __restrict__ k,
                                                   const __bf16* __restrict__ vT,
                                                   const int* __restrict__ mask,
                                                   __bf16* __restrict__ attn_out) {
    __shared__ __bf16 lds[4][16 * 32];
    int wave = threadIdx.x >> 5;
    int lane = threadIdx.x & 31;
    int lo = lane & 15, hi = lane >> 4, kb = hi * 8;
    int gw = blockIdx.x * 4 + wave;        // 0 .. B*H*(L/16)-1
    int qt = gw % (Lq / 16);
    int bh = gw / (Lq / 16);
    int h = bh % Hh, b = bh / Hh;

    const float scale = 0.125f;                       // 1/sqrt(64)
    const float slope = exp2f(-(float)(h + 1));       // 2^(-8*(h+1)/H), H=8

    // Q fragments (two K-chunks of 32 covering HD=64)
    const __bf16* qbase = q + ((size_t)(b * Lq + qt * 16 + lo)) * Dm + h * HD;
    v16bf aq0 = ldfrag(qbase + kb);
    v16bf aq1 = ldfrag(qbase + 32 + kb);

    v8f o0 = {}, o1 = {}, o2 = {}, o3 = {};
    float rmax[8], rsum[8], qpos[8];
#pragma unroll
    for (int r = 0; r < 8; ++r) {
        rmax[r] = -1e30f; rsum[r] = 0.f;
        qpos[r] = (float)(qt * 16 + r + 8 * hi);
    }

    for (int kt = 0; kt < Lq / 32; ++kt) {
        int n0 = kt * 32;

        // ---- hoisted loads: 4 K-fragments + 4 V-fragments + mask ----
        const __bf16* kr0 = k + ((size_t)(b * Lq + n0 + lo)) * Dm + h * HD + kb;
        const __bf16* kr1 = kr0 + (size_t)16 * Dm;
        v16bf k00 = ldfrag(kr0);
        v16bf k01 = ldfrag(kr0 + 32);
        v16bf k10 = ldfrag(kr1);
        v16bf k11 = ldfrag(kr1 + 32);

        const __bf16* vb = vT + ((size_t)(b * Hh + h) * HD) * Lq + n0 + kb;
        v16bf bv0 = ldfrag(vb + (size_t)(0 * 16 + lo) * Lq);
        v16bf bv1 = ldfrag(vb + (size_t)(1 * 16 + lo) * Lq);
        v16bf bv2 = ldfrag(vb + (size_t)(2 * 16 + lo) * Lq);
        v16bf bv3 = ldfrag(vb + (size_t)(3 * 16 + lo) * Lq);

        bool msk0 = mask[b * Lq + n0 + lo] != 0;
        bool msk1 = mask[b * Lq + n0 + 16 + lo] != 0;

        // ---- scores: S[16,32] = Q[16,64] . K^T ----
        v8f s0 = {}, s1 = {};
        s0 = wmma_bf16(aq0, k00, s0);
        s0 = wmma_bf16(aq1, k01, s0);
        s1 = wmma_bf16(aq0, k10, s1);
        s1 = wmma_bf16(aq1, k11, s1);

        float kp0 = (float)(n0 + lo);
        float kp1 = (float)(n0 + 16 + lo);
#pragma unroll
        for (int r = 0; r < 8; ++r) {
            float v0 = s0[r] * scale - slope * fabsf(qpos[r] - kp0);
            float v1 = s1[r] * scale - slope * fabsf(qpos[r] - kp1);
            s0[r] = msk0 ? -1e30f : v0;
            s1[r] = msk1 ? -1e30f : v1;
        }

        // ---- online softmax (rows striped across the 16 lanes of each half) ----
        float alpha[8];
#pragma unroll
        for (int r = 0; r < 8; ++r) {
            float t = fmaxf(s0[r], s1[r]);
#pragma unroll
            for (int m = 1; m < 16; m <<= 1) t = fmaxf(t, __shfl_xor(t, m, 16));
            float mnew = fmaxf(rmax[r], t);
            alpha[r] = __expf(rmax[r] - mnew);
            rmax[r] = mnew;
            float p0 = __expf(s0[r] - mnew);
            float p1 = __expf(s1[r] - mnew);
            s0[r] = p0; s1[r] = p1;
            float ps = p0 + p1;
#pragma unroll
            for (int m = 1; m < 16; m <<= 1) ps += __shfl_xor(ps, m, 16);
            rsum[r] = rsum[r] * alpha[r] + ps;
        }
#pragma unroll
        for (int r = 0; r < 8; ++r) {
            o0[r] *= alpha[r]; o1[r] *= alpha[r];
            o2[r] *= alpha[r]; o3[r] *= alpha[r];
        }

        // ---- transpose P (16x32) through LDS into an A-fragment ----
        __bf16* ld = lds[wave];
#pragma unroll
        for (int r = 0; r < 8; ++r) {
            int row = r + 8 * hi;
            ld[row * 32 + lo]      = (__bf16)s0[r];
            ld[row * 32 + 16 + lo] = (__bf16)s1[r];
        }
        asm volatile("s_wait_dscnt 0" ::: "memory");
        v16bf ap = ldfrag(ld + lo * 32 + kb);

        // ---- O[16,64] += P[16,32] @ V[32,64] (V pre-loaded, transposed [d][l]) ----
        o0 = wmma_bf16(ap, bv0, o0);
        o1 = wmma_bf16(ap, bv1, o1);
        o2 = wmma_bf16(ap, bv2, o2);
        o3 = wmma_bf16(ap, bv3, o3);
    }

#pragma unroll
    for (int r = 0; r < 8; ++r) {
        int m = qt * 16 + r + 8 * hi;
        float inv = 1.0f / rsum[r];
        size_t base = (size_t)(b * Lq + m) * Dm + h * HD;
        attn_out[base + 0 * 16 + lo] = (__bf16)(o0[r] * inv);
        attn_out[base + 1 * 16 + lo] = (__bf16)(o1[r] * inv);
        attn_out[base + 2 * 16 + lo] = (__bf16)(o2[r] * inv);
        attn_out[base + 3 * 16 + lo] = (__bf16)(o3[r] * inv);
    }
}

// ---------------- host orchestration ----------------
extern "C" void kernel_launch(void* const* d_in, const int* in_sizes, int n_in,
                              void* d_out, int out_size, void* d_ws, size_t ws_size,
                              hipStream_t stream) {
    const float* src  = (const float*)d_in[0];
    const float* tgt  = (const float*)d_in[1];
    // d_in[2] = alibi_bias (computed analytically instead)
    const float* pos  = (const float*)d_in[3];
    const int*   mask = (const int*)d_in[4];
    const float* Wl = (const float*)d_in[5];  const float* bl = (const float*)d_in[6];
    const float* Wq = (const float*)d_in[7];  const float* bq = (const float*)d_in[8];
    const float* Wk = (const float*)d_in[9];  const float* bk = (const float*)d_in[10];
    const float* Wv = (const float*)d_in[11]; const float* bv = (const float*)d_in[12];
    const float* Wo = (const float*)d_in[13]; const float* bo = (const float*)d_in[14];
    float* out = (float*)d_out;

    char* ws = (char*)d_ws;
    size_t cur = 0;
    auto alloc = [&](size_t bytes) {
        void* p = ws + cur;
        cur = (cur + bytes + 255) & ~(size_t)255;
        return p;
    };
    __bf16* src_bf  = (__bf16*)alloc((size_t)ML * Dm * 2);
    __bf16* tgt_bf  = (__bf16*)alloc((size_t)ML * Dm * 2);
    __bf16* WlT     = (__bf16*)alloc((size_t)Pp * Dm * 2);
    __bf16* WqT     = (__bf16*)alloc((size_t)Dm * Dm * 2);
    __bf16* WkT     = (__bf16*)alloc((size_t)Dm * Pp * 2);
    __bf16* WvT     = (__bf16*)alloc((size_t)Dm * Pp * 2);
    __bf16* WoT     = (__bf16*)alloc((size_t)Dm * Dm * 2);
    __bf16* proj_bf = (__bf16*)alloc((size_t)ML * Pp * 2);
    float*  tmpf    = (float*)alloc((size_t)ML * Dm * 4);
    __bf16* q_bf    = (__bf16*)alloc((size_t)ML * Dm * 2);
    __bf16* k_bf    = (__bf16*)alloc((size_t)ML * Dm * 2);
    __bf16* v_bf    = (__bf16*)alloc((size_t)ML * Dm * 2);
    __bf16* vT_bf   = (__bf16*)alloc((size_t)ML * Dm * 2);
    __bf16* attn_bf = (__bf16*)alloc((size_t)ML * Dm * 2);

    const int TB = 256;
    int nel = ML * Dm;
    cvt_bf16_kernel<<<(nel + TB - 1) / TB, TB, 0, stream>>>(src, src_bf, nel);
    cvt_bf16_kernel<<<(nel + TB - 1) / TB, TB, 0, stream>>>(tgt, tgt_bf, nel);

    transpose_w_kernel<<<(Dm * Pp + TB - 1) / TB, TB, 0, stream>>>(Wl, WlT, Dm, Pp);
    transpose_w_kernel<<<(Dm * Dm + TB - 1) / TB, TB, 0, stream>>>(Wq, WqT, Dm, Dm);
    transpose_w_kernel<<<(Pp * Dm + TB - 1) / TB, TB, 0, stream>>>(Wk, WkT, Pp, Dm);
    transpose_w_kernel<<<(Pp * Dm + TB - 1) / TB, TB, 0, stream>>>(Wv, WvT, Pp, Dm);
    transpose_w_kernel<<<(Dm * Dm + TB - 1) / TB, TB, 0, stream>>>(Wo, WoT, Dm, Dm);

    auto gemm = [&](const __bf16* A, const __bf16* Bt, const float* bias,
                    float* Cf, __bf16* Cb, int M, int N, int K) {
        int tiles = (M / 16) * (N / 64);
        int blocks = (tiles + 7) / 8;       // 8 waves per 256-thread block
        gemm_bf16_kernel<<<blocks, 256, 0, stream>>>(A, Bt, bias, Cf, Cb, M, N, K);
    };

    // proj = src @ Wl + bl   -> bf16
    gemm(src_bf, WlT, bl, nullptr, proj_bf, ML, Pp, Dm);
    // q = tgt @ Wq + bq      -> f32, then RoPE -> bf16
    gemm(tgt_bf, WqT, bq, tmpf, nullptr, ML, Dm, Dm);
    rope_kernel<<<(ML * Hh * 32 + TB - 1) / TB, TB, 0, stream>>>(tmpf, pos, q_bf);
    // k = proj @ Wk + bk     -> f32, then RoPE -> bf16
    gemm(proj_bf, WkT, bk, tmpf, nullptr, ML, Dm, Pp);
    rope_kernel<<<(ML * Hh * 32 + TB - 1) / TB, TB, 0, stream>>>(tmpf, pos, k_bf);
    // v = proj @ Wv + bv     -> bf16, then transpose to [b,h,d,l]
    gemm(proj_bf, WvT, bv, nullptr, v_bf, ML, Dm, Pp);
    transpose_v_kernel<<<(ML * Dm + TB - 1) / TB, TB, 0, stream>>>(v_bf, vT_bf);

    // fused attention: B*H*(L/16) waves, 4 waves per block
    int awaves = Bb * Hh * (Lq / 16);
    attn_kernel<<<awaves / 4, 128, 0, stream>>>(q_bf, k_bf, vT_bf, mask, attn_bf);

    // out = attn @ Wo + bo   -> f32 into d_out
    gemm(attn_bf, WoT, bo, out, nullptr, ML, Dm, Dm);

    (void)in_sizes; (void)n_in; (void)out_size; (void)ws_size;
}
